// GraphSAGEModel_2001454760098
// MI455X (gfx1250) — matrix operand order repaired and checked by reading, not compiled, not run on hardware
//
#include <hip/hip_runtime.h>

// ---------------------------------------------------------------------------
// GraphSAGE (2x SAGEConv, mean aggregation) for gfx1250 / MI455X.
//   - GEMMs: v_wmma_f32_16x16x32_bf16, wave32, 16x16 tile/wave, K fully unrolled
//   - Edge aggregation: native global_atomic_add_f32 (inline asm, no-return)
//   - Working set (x 25.6MB, agg 51.2MB, h 51.2MB) is L2-resident (192MB L2)
// ---------------------------------------------------------------------------

typedef __attribute__((ext_vector_type(16))) __bf16 v16bf;
typedef __attribute__((ext_vector_type(8)))  float  v8f;

#define NNODES 50000
#define MTILES (NNODES / 16)   // 3125, exact

// Guaranteed-native fp32 global atomic add, no return value (STOREcnt path).
__device__ __forceinline__ void atomic_add_f32(float* p, float v) {
  asm volatile("global_atomic_add_f32 %0, %1, off"
               :
               : "v"(p), "v"(v)
               : "memory");
}

// ---------------------------------------------------------------------------
__global__ void zero_f32(float* __restrict__ p, size_t n) {
  size_t i = (size_t)blockIdx.x * blockDim.x + threadIdx.x;
  if (i < n) p[i] = 0.0f;
}

// deg[dst] += 1 per edge
__global__ void degree_kernel(const long long* __restrict__ dst, int E,
                              float* __restrict__ deg) {
  int e = blockIdx.x * blockDim.x + threadIdx.x;
  if (e < E) atomic_add_f32(&deg[dst[e]], 1.0f);
}

// agg[dst] += x[src], D floats per edge, 4 floats per thread
template <int D>
__global__ void scatter_accum(const float* __restrict__ x,
                              const long long* __restrict__ src,
                              const long long* __restrict__ dst,
                              int E, float* __restrict__ agg) {
  constexpr int LPE = D / 4;  // lanes per edge (power of two -> shifts)
  size_t tid = (size_t)blockIdx.x * blockDim.x + threadIdx.x;
  size_t e = tid / LPE;
  int c = (int)(tid % LPE) * 4;
  if (e >= (size_t)E) return;
  long long s = src[e];
  long long d = dst[e];
  const float4 v = *(const float4*)(x + (size_t)s * D + c);
  float* a = agg + (size_t)d * D + c;
  atomic_add_f32(a + 0, v.x);
  atomic_add_f32(a + 1, v.y);
  atomic_add_f32(a + 2, v.z);
  atomic_add_f32(a + 3, v.w);
}

// ---------------------------------------------------------------------------
// Pack A-operand [NNODES x K] f32 into WMMA bf16 A-fragment layout.
// Fragment unit = (mtile, ktile, lane): 16 bf16 stored contiguously at
//   Ap[((mtile*Kt + ktile)*32 + lane)*16].
// ISA 16-bit A 16x32 layout: lanes 0-15 -> K {0..7,16..23},
// lanes 16-31 -> K {8..15,24..31}; row = lane & 15.
// Optional per-row scale = 1/max(deg,1) implements the mean aggregation.
__global__ void pack_a_bf16(const float* __restrict__ A,
                            const float* __restrict__ deg,   // may be null
                            unsigned short* __restrict__ Ap,
                            int Kt, int K) {
  size_t tid = (size_t)blockIdx.x * blockDim.x + threadIdx.x;
  size_t nthreads = (size_t)MTILES * Kt * 32;
  if (tid >= nthreads) return;
  int lane = (int)(tid & 31);
  size_t t = tid >> 5;
  int kt = (int)(t % Kt);
  int mt = (int)(t / Kt);
  int row = mt * 16 + (lane & 15);
  int kb  = kt * 32 + ((lane & 16) ? 8 : 0);
  float scale = 1.0f;
  if (deg) scale = 1.0f / fmaxf(deg[row], 1.0f);
  const float* s = A + (size_t)row * K + kb;
  v16bf v;
#pragma unroll
  for (int j = 0; j < 8; ++j) v[j]     = (__bf16)(s[j]      * scale);
#pragma unroll
  for (int j = 0; j < 8; ++j) v[j + 8] = (__bf16)(s[16 + j] * scale);
  *(v16bf*)(Ap + tid * 16) = v;
}

// Pack weight [K x N] f32 (row-major) into WMMA bf16 B-fragment layout.
// Fragment unit = (ktile, ntile, lane): Wp[((ktile*Nt + ntile)*32 + lane)*16].
// ISA 16-bit B 32x16 layout: col = lane & 15; lanes 0-15 hold K {0..7,16..23},
// lanes 16-31 hold K {8..15,24..31} of the k-tile.
__global__ void pack_w_bf16(const float* __restrict__ W,
                            unsigned short* __restrict__ Wp,
                            int Kt, int Nt, int N) {
  int tid = blockIdx.x * blockDim.x + threadIdx.x;
  int nthreads = Kt * Nt * 32;
  if (tid >= nthreads) return;
  int lane = tid & 31;
  int t = tid >> 5;
  int nt = t % Nt;
  int kt = t / Nt;
  int col = nt * 16 + (lane & 15);
  int kb  = kt * 32 + ((lane & 16) ? 8 : 0);
  v16bf v;
#pragma unroll
  for (int j = 0; j < 16; ++j) v[j] = (__bf16)W[(size_t)(kb + j + (j >= 8 ? 8 : 0)) * N + col];
  *(v16bf*)(Wp + (size_t)tid * 16) = v;
}

// ---------------------------------------------------------------------------
// Fused SAGE layer GEMM: out = [relu]( Am@Wl + Ax@Wr + bias )
// One wave computes one 16x16 output tile; 8 waves/block span 8 N-tiles.
// KT/NT compile-time -> K loop fully unrolled, long load clauses.
template <int KT, int NT>
__global__ void sage_gemm_wmma(const unsigned short* __restrict__ Am,
                               const unsigned short* __restrict__ Wl,
                               const unsigned short* __restrict__ Ax,
                               const unsigned short* __restrict__ Wr,
                               const float* __restrict__ bias,
                               float* __restrict__ out,
                               int N, int relu) {
  int lane = threadIdx.x & 31;
  int wave = threadIdx.x >> 5;
  int mt = blockIdx.x;
  int nt = blockIdx.y * (blockDim.x >> 5) + wave;
  if (nt >= NT) return;  // wave-uniform: EXEC stays all-1s for active waves

  const unsigned short* a1 = Am + (size_t)mt * KT * 512 + (size_t)lane * 16;
  const unsigned short* a2 = Ax + (size_t)mt * KT * 512 + (size_t)lane * 16;
  const unsigned short* b1 = Wl + (size_t)nt * 512 + (size_t)lane * 16;
  const unsigned short* b2 = Wr + (size_t)nt * 512 + (size_t)lane * 16;

  v8f acc = {};
#pragma unroll
  for (int kt = 0; kt < KT; ++kt) {
    v16bf va1 = *(const v16bf*)(a1 + (size_t)kt * 512);
    v16bf vb1 = *(const v16bf*)(b1 + (size_t)kt * NT * 512);
    acc = __builtin_amdgcn_wmma_f32_16x16x32_bf16(false, va1, false, vb1,
                                                  (short)0, acc, false, false);
    v16bf va2 = *(const v16bf*)(a2 + (size_t)kt * 512);
    v16bf vb2 = *(const v16bf*)(b2 + (size_t)kt * NT * 512);
    acc = __builtin_amdgcn_wmma_f32_16x16x32_bf16(false, va2, false, vb2,
                                                  (short)0, acc, false, false);
    if (kt + 1 < KT) {
      // locality=3 -> WGP-scope prefetch (fills all cache levels)
      __builtin_prefetch(a1 + (size_t)(kt + 1) * 512, 0, 3);
      __builtin_prefetch(a2 + (size_t)(kt + 1) * 512, 0, 3);
    }
  }

  float bv = bias ? bias[nt * 16 + (lane & 15)] : 0.0f;
  int rowbase = mt * 16 + ((lane & 16) ? 8 : 0);
  int col = nt * 16 + (lane & 15);
#pragma unroll
  for (int r = 0; r < 8; ++r) {
    float v = acc[r] + bv;
    if (relu) v = fmaxf(v, 0.0f);
    out[(size_t)(rowbase + r) * N + col] = v;
  }
}

// ---------------------------------------------------------------------------
extern "C" void kernel_launch(void* const* d_in, const int* in_sizes, int n_in,
                              void* d_out, int out_size, void* d_ws, size_t ws_size,
                              hipStream_t stream) {
  const float*     x   = (const float*)d_in[0];
  const long long* ei  = (const long long*)d_in[1];   // int64 edge_index [2, E]
  const float*     Wl1 = (const float*)d_in[2];
  const float*     bl1 = (const float*)d_in[3];
  const float*     Wr1 = (const float*)d_in[4];
  const float*     Wl2 = (const float*)d_in[5];
  const float*     bl2 = (const float*)d_in[6];
  const float*     Wr2 = (const float*)d_in[7];

  const int E = in_sizes[1] / 2;
  const long long* src = ei;
  const long long* dst = ei + E;

  // ---- workspace carve-up (256B aligned regions) ----
  char* ws = (char*)d_ws;
  size_t off = 0;
  auto take = [&](size_t bytes) -> void* {
    void* p = ws + off;
    off += (bytes + 255) & ~(size_t)255;
    return p;
  };
  float* deg = (float*)take((size_t)NNODES * 4);             //  0.2 MB
  float* agg = (float*)take((size_t)NNODES * 256 * 4);       // 51.2 MB (both layers)
  float* h   = (float*)take((size_t)NNODES * 256 * 4);       // 51.2 MB
  unsigned short* apx = (unsigned short*)take((size_t)NNODES * 256 * 2); // x / h pack
  unsigned short* apm = (unsigned short*)take((size_t)NNODES * 256 * 2); // mean pack
  unsigned short* wp1l = (unsigned short*)take((size_t)128 * 256 * 2);
  unsigned short* wp1r = (unsigned short*)take((size_t)128 * 256 * 2);
  unsigned short* wp2l = (unsigned short*)take((size_t)256 * 128 * 2);
  unsigned short* wp2r = (unsigned short*)take((size_t)256 * 128 * 2);

  const int B = 256;
  auto blocks = [](size_t n, int b) { return (unsigned)((n + b - 1) / b); };

  // ---- prepack weights (tiny) ----
  pack_w_bf16<<<blocks(4 * 16 * 32, B), B, 0, stream>>>(Wl1, wp1l, 4, 16, 256);
  pack_w_bf16<<<blocks(4 * 16 * 32, B), B, 0, stream>>>(Wr1, wp1r, 4, 16, 256);
  pack_w_bf16<<<blocks(8 * 8 * 32, B), B, 0, stream>>>(Wl2, wp2l, 8, 8, 128);
  pack_w_bf16<<<blocks(8 * 8 * 32, B), B, 0, stream>>>(Wr2, wp2r, 8, 8, 128);

  // ---- degree (shared by both layers) ----
  zero_f32<<<blocks(NNODES, B), B, 0, stream>>>(deg, NNODES);
  degree_kernel<<<blocks(E, B), B, 0, stream>>>(dst, E, deg);

  // ================= layer 1 (128 -> 256, relu) =================
  {
    size_t naz = (size_t)NNODES * 128;
    zero_f32<<<blocks(naz, B), B, 0, stream>>>(agg, naz);
    size_t nthr = (size_t)E * 32;  // 128/4 lanes per edge
    scatter_accum<128><<<blocks(nthr, B), B, 0, stream>>>(x, src, dst, E, agg);

    size_t npk = (size_t)MTILES * 4 * 32;
    pack_a_bf16<<<blocks(npk, B), B, 0, stream>>>(x, nullptr, apx, 4, 128);
    pack_a_bf16<<<blocks(npk, B), B, 0, stream>>>(agg, deg, apm, 4, 128);

    dim3 g(MTILES, 2);  // 16 N-tiles / 8 waves per block
    sage_gemm_wmma<4, 16><<<g, B, 0, stream>>>(apm, wp1l, apx, wp1r, bl1, h,
                                               /*N=*/256, /*relu=*/1);
  }

  // ================= layer 2 (256 -> 128) =================
  {
    size_t naz = (size_t)NNODES * 256;
    zero_f32<<<blocks(naz, B), B, 0, stream>>>(agg, naz);
    size_t nthr = (size_t)E * 64;  // 256/4 lanes per edge
    scatter_accum<256><<<blocks(nthr, B), B, 0, stream>>>(h, src, dst, E, agg);

    size_t npk = (size_t)MTILES * 8 * 32;
    pack_a_bf16<<<blocks(npk, B), B, 0, stream>>>(h, nullptr, apx, 8, 256);
    pack_a_bf16<<<blocks(npk, B), B, 0, stream>>>(agg, deg, apm, 8, 256);

    dim3 g(MTILES, 1);  // 8 N-tiles / 8 waves per block
    sage_gemm_wmma<8, 8><<<g, B, 0, stream>>>(apm, wp2l, apx, wp2r, bl2, (float*)d_out,
                                              /*N=*/128, /*relu=*/0);
  }
}
